// LMModel_LSTM_22823456211735
// MI455X (gfx1250) — matrix operand order repaired and checked by reading, not compiled, not run on hardware
//
#include <hip/hip_runtime.h>

// ---------------------------------------------------------------------------
// Types for CDNA5 WMMA (wave32): v_wmma_f32_16x16x32_bf16
// ---------------------------------------------------------------------------
typedef __bf16 bf16_t;
typedef __attribute__((ext_vector_type(16))) __bf16 v16bf;
typedef __attribute__((ext_vector_type(8)))  __bf16 v8bf;
typedef __attribute__((ext_vector_type(8)))  float  v8f;

#define SEQ    128
#define BATCH  32
#define HID    1024
#define NVOC   32000
#define NGATE  4

// ---------------------------------------------------------------------------
// WMMA fragment loaders. A and B are bf16, K-contiguous ("NT" gemm):
//   A: row-major [M][K]   (rows = batch / timestep-rows)
//   B: row-major [N][K]   (rows = output columns; weights pre-transposed)
// Per-lane layouts per CDNA5 ISA §7.12.2 (16-bit A 16x32, B 32x16).
// ---------------------------------------------------------------------------
__device__ __forceinline__ v16bf load_frag_a(const bf16_t* __restrict__ base,
                                             int ldk, int m0, int k0) {
  const int lane = threadIdx.x & 31;
  const int row  = m0 + (lane & 15);
  const int kb   = k0 + ((lane & 16) ? 8 : 0);     // lanes 0-15: K+0/K+16, 16-31: K+8/K+24
  const bf16_t* p = base + (size_t)row * ldk + kb;
  union { v16bf v; v8bf h[2]; } u;
  u.h[0] = *(const v8bf*)(p);
  u.h[1] = *(const v8bf*)(p + 16);
  return u.v;
}

__device__ __forceinline__ v16bf load_frag_b(const bf16_t* __restrict__ base,
                                             int ldk, int n0, int k0) {
  const int lane = threadIdx.x & 31;
  const int row  = n0 + (lane & 15);               // column of B == row of [N][K] array
  const int kb   = k0 + ((lane & 16) ? 8 : 0);     // lanes 0-15: K+0/K+16, 16-31: K+8/K+24
  const bf16_t* p = base + (size_t)row * ldk + kb;
  union { v16bf v; v8bf h[2]; } u;
  u.h[0] = *(const v8bf*)(p);
  u.h[1] = *(const v8bf*)(p + 16);
  return u.v;
}

__device__ __forceinline__ v8f wmma_bf16(v16bf a, v16bf b, v8f c) {
  return __builtin_amdgcn_wmma_f32_16x16x32_bf16(false, a, false, b, (short)0, c,
                                                 false, false);
}

__device__ __forceinline__ float sigf(float x) {
  return 1.0f / (1.0f + __expf(-x));
}

// ---------------------------------------------------------------------------
// Device-wide split barrier (persistent kernel, 64 co-resident blocks).
// ---------------------------------------------------------------------------
__device__ __forceinline__ void gbar(unsigned* counter, unsigned phase) {
  __threadfence();
  __syncthreads();
  if (threadIdx.x == 0) {
    atomicAdd(counter, 1u);
    const unsigned target = phase * gridDim.x;
    while (__hip_atomic_load(counter, __ATOMIC_ACQUIRE, __HIP_MEMORY_SCOPE_AGENT) < target) {
      __builtin_amdgcn_s_sleep(2);
    }
  }
  __syncthreads();
}

// ---------------------------------------------------------------------------
// fp32 [K][N] -> bf16 [N][K] tiled transpose-convert (32x32 tiles in LDS).
// ---------------------------------------------------------------------------
__global__ void transpose_cvt(const float* __restrict__ src, bf16_t* __restrict__ dst,
                              int K, int N) {
  __shared__ float tile[32][33];
  const int n0 = blockIdx.x * 32;
  const int k0 = blockIdx.y * 32;
  const int tx = threadIdx.x & 31;
  const int ty = threadIdx.x >> 5;   // 0..7
#pragma unroll
  for (int i = 0; i < 32; i += 8)
    tile[ty + i][tx] = src[(size_t)(k0 + ty + i) * N + (n0 + tx)];
  __syncthreads();
#pragma unroll
  for (int i = 0; i < 32; i += 8)
    dst[(size_t)(n0 + ty + i) * K + (k0 + tx)] = (bf16_t)tile[tx][ty + i];
}

__global__ void cvt_bf16(const float* __restrict__ src, bf16_t* __restrict__ dst,
                         size_t n) {
  size_t i = (size_t)blockIdx.x * blockDim.x + threadIdx.x;
  const size_t stride = (size_t)gridDim.x * blockDim.x;
  for (; i < n; i += stride) dst[i] = (bf16_t)src[i];
}

// Embedding gather + fp32->bf16: one block per (t,b) row.
__global__ void embed_cvt(const int* __restrict__ ids, const float* __restrict__ emb,
                          bf16_t* __restrict__ xbf) {
  const int row = blockIdx.x;                 // 0..4095 = t*32+b
  const int id  = ids[row];
  const float* s = emb + (size_t)id * HID;
  bf16_t* d = xbf + (size_t)row * HID;
  for (int k = threadIdx.x; k < HID; k += blockDim.x) d[k] = (bf16_t)s[k];
}

// ---------------------------------------------------------------------------
// bf16 WMMA GEMM, C[M][N] f32 = A[M][K] * B[N][K]^T (+ bias[n]).
// 256 threads = 8 waves, wave grid 2(M) x 4(N); wave tile 32x64
// (2 M-tiles x 4 N-tiles = 8 WMMA per k-step from 6 fragment loads
//  -> ~22 FLOP/B cache intensity). Block tile 64 x 256.
// STREAM_C: non-temporal stores so the 524MB logits stream does not
// evict the L2-resident bf16 decoder weights.
// ---------------------------------------------------------------------------
template <bool STREAM_C>
__global__ void gemm_bf16_nt(const bf16_t* __restrict__ A, const bf16_t* __restrict__ B,
                             const float* __restrict__ bias, float* __restrict__ C,
                             int N, int K) {
  const int wave = threadIdx.x >> 5;
  const int lane = threadIdx.x & 31;
  const int m0 = blockIdx.x * 64 + (wave & 1) * 32;    // wave covers rows m0..m0+31
  const int n0 = blockIdx.y * 256 + (wave >> 1) * 64;  // wave covers cols n0..n0+63

  v8f acc[2][4] = {};
  for (int k0 = 0; k0 < K; k0 += 32) {
    if (k0 + 128 < K) {  // prefetch B stream one block-K ahead (global_prefetch_b8)
      __builtin_prefetch(&B[(size_t)(n0 + (lane & 15)) * K + k0 + 128], 0, 1);
      __builtin_prefetch(&B[(size_t)(n0 + 32 + (lane & 15)) * K + k0 + 128], 0, 1);
    }
    v16bf a0 = load_frag_a(A, K, m0, k0);
    v16bf a1 = load_frag_a(A, K, m0 + 16, k0);
    v16bf b0 = load_frag_b(B, K, n0, k0);
    v16bf b1 = load_frag_b(B, K, n0 + 16, k0);
    v16bf b2 = load_frag_b(B, K, n0 + 32, k0);
    v16bf b3 = load_frag_b(B, K, n0 + 48, k0);
    acc[0][0] = wmma_bf16(a0, b0, acc[0][0]);
    acc[0][1] = wmma_bf16(a0, b1, acc[0][1]);
    acc[0][2] = wmma_bf16(a0, b2, acc[0][2]);
    acc[0][3] = wmma_bf16(a0, b3, acc[0][3]);
    acc[1][0] = wmma_bf16(a1, b0, acc[1][0]);
    acc[1][1] = wmma_bf16(a1, b1, acc[1][1]);
    acc[1][2] = wmma_bf16(a1, b2, acc[1][2]);
    acc[1][3] = wmma_bf16(a1, b3, acc[1][3]);
  }

  const int colb = lane & 15;
  const int rbo  = (lane & 16) ? 8 : 0;
#pragma unroll
  for (int mt = 0; mt < 2; ++mt) {
#pragma unroll
    for (int nt = 0; nt < 4; ++nt) {
      const int col = n0 + nt * 16 + colb;
      const float bv = bias ? bias[col] : 0.0f;
      float* p = C + (size_t)(m0 + mt * 16 + rbo) * N + col;
#pragma unroll
      for (int r = 0; r < 8; ++r) {
        const float v = acc[mt][nt][r] + bv;
        if (STREAM_C) __builtin_nontemporal_store(v, p);
        else          *p = v;
        p += N;
      }
    }
  }
}

// ---------------------------------------------------------------------------
// One LSTM layer micro-step for this block's 16-column slice.
// B-fragments come from the LDS-resident weight slices (staged once for all
// 128 steps). K-loops kept partially unrolled (8) so the compiler cannot
// hoist all 32 k-steps of fragments across the t-loop (round-2 did that and
// spilled to scratch inside the serial recurrence).
// ---------------------------------------------------------------------------
__device__ __forceinline__ void lstm_layer(
    const bf16_t* __restrict__ hread,        // [32][1024] bf16 (prev h, this layer)
    const bf16_t* __restrict__ inp,          // [32][1024] bf16 (layer input) or nullptr
    const float*  __restrict__ XuRow,        // &Xu[t*32*4096] or nullptr (layer 0)
    const bf16_t* sW,                        // LDS [64][1024]: rows g*16+nn of W slice
    const bf16_t* sU,                        // LDS [64][1024]: rows g*16+nn of U slice
    const float*  __restrict__ bias_g,       // bias for this wave's gate
    float* __restrict__ cstate,              // [32][1024] f32 (owner slice only)
    bf16_t* __restrict__ hwrite,             // [32][1024] bf16
    bf16_t* __restrict__ outs_row,           // &outs[t*32*1024] or nullptr
    float sg[NGATE][32][16], int n0, int m_t, int g) {
  v8f acc = {};
#pragma unroll 8
  for (int k0 = 0; k0 < HID; k0 += 32) {
    v16bf a = load_frag_a(hread, HID, m_t * 16, k0);
    v16bf b = load_frag_b(sW, HID, g * 16, k0);    // ds_load from LDS slice
    acc = wmma_bf16(a, b, acc);
  }
  if (inp) {  // layer 1: + inp @ U (uniform branch, EXEC stays all-ones)
#pragma unroll 8
    for (int k0 = 0; k0 < HID; k0 += 32) {
      v16bf a = load_frag_a(inp, HID, m_t * 16, k0);
      v16bf b = load_frag_b(sU, HID, g * 16, k0);
      acc = wmma_bf16(a, b, acc);
    }
  }
  const int lane = threadIdx.x & 31;
  const int col  = lane & 15;
  const int rb   = m_t * 16 + ((lane & 16) ? 8 : 0);
  const float bv = bias_g[n0 + col];
#pragma unroll
  for (int r = 0; r < 8; ++r) {
    float pre = acc[r] + bv;
    if (XuRow)  // layer 0: precomputed x@U term
      pre += XuRow[(size_t)(rb + r) * (NGATE * HID) + g * HID + n0 + col];
    sg[g][rb + r][col] = pre;
  }
  __syncthreads();
  for (int e = threadIdx.x; e < 32 * 16; e += blockDim.x) {
    const int m = e >> 4, nn = e & 15;
    const float fg = sigf(sg[0][m][nn]);
    const float ig = sigf(sg[1][m][nn]);
    const float gg = tanhf(sg[2][m][nn]);
    const float og = sigf(sg[3][m][nn]);
    const float cp = cstate[m * HID + n0 + nn];
    const float cn = fg * cp + ig * gg;
    cstate[m * HID + n0 + nn] = cn;
    const float hn = og * tanhf(cn);
    hwrite[m * HID + n0 + nn] = (bf16_t)hn;
    if (outs_row) outs_row[m * HID + n0 + nn] = (bf16_t)hn;
  }
}

// ---------------------------------------------------------------------------
// Persistent recurrent kernel: 64 blocks x 256 threads, one block per WGP.
// Block b owns h-columns [16b, 16b+16) for all 4 gates. Dynamic LDS:
//   [0,128K)   W slice  (64 rows x 1024 K bf16)
//   [128K,256K) U slice
//   [256K,264K) gate-preact exchange buffer (f32[4][32][16])
// Weights are read from global exactly once; the 256 serial phases then run
// entirely out of LDS (+ L2-hot h vectors). Global split-barrier per phase.
// ---------------------------------------------------------------------------
__global__ void lstm_persistent(const bf16_t* __restrict__ Wt, const bf16_t* __restrict__ Ut,
                                const float* __restrict__ Xu,
                                const float* __restrict__ bfp, const float* __restrict__ bip,
                                const float* __restrict__ bcp, const float* __restrict__ bop,
                                bf16_t* __restrict__ h0buf, bf16_t* __restrict__ h1buf,
                                float* __restrict__ c0, float* __restrict__ c1,
                                bf16_t* __restrict__ outs, unsigned* counter) {
  extern __shared__ char ldsbuf[];
  bf16_t* sW = (bf16_t*)ldsbuf;                                   // [64][1024]
  bf16_t* sU = (bf16_t*)(ldsbuf + (size_t)64 * HID * 2);          // [64][1024]
  float (*sg)[32][16] = (float (*)[32][16])(ldsbuf + (size_t)2 * 64 * HID * 2);

  const int n0   = blockIdx.x * 16;
  const int wave = threadIdx.x >> 5;
  const int m_t  = wave & 1;
  const int g    = wave >> 1;
  const float* bias_g = (g == 0) ? bfp : (g == 1) ? bip : (g == 2) ? bcp : bop;

  // ---- stage this block's W/U slices into LDS (16B vector copies) ----
  for (int idx = threadIdx.x; idx < 64 * (HID / 8); idx += blockDim.x) {
    const int row = idx >> 7;              // 0..63 = gate*16 + nn
    const int ch  = idx & 127;             // 8-element chunk within the row
    const int gg  = row >> 4, nn = row & 15;
    const size_t grow = (size_t)(gg * HID + n0 + nn) * HID + ch * 8;
    *(v8bf*)(sW + (size_t)row * HID + ch * 8) = *(const v8bf*)(Wt + grow);
    *(v8bf*)(sU + (size_t)row * HID + ch * 8) = *(const v8bf*)(Ut + grow);
  }
  __syncthreads();

  unsigned phase = 0;
  const int HB = BATCH * HID;  // one h buffer

  for (int t = 0; t < SEQ; ++t) {
    const int p = t & 1;
    // layer 0: gates = h0_prev @ W + Xu[t] + b
    lstm_layer(h0buf + p * HB, nullptr, Xu + (size_t)t * BATCH * NGATE * HID,
               sW, sU, bias_g, c0, h0buf + (1 - p) * HB, nullptr, sg, n0, m_t, g);
    gbar(counter, ++phase);
    // layer 1: gates = h1_prev @ W + h0_new @ U + b ; emits outs[t]
    lstm_layer(h1buf + p * HB, h0buf + (1 - p) * HB, nullptr,
               sW, sU, bias_g, c1, h1buf + (1 - p) * HB,
               outs + (size_t)t * BATCH * HID, sg, n0, m_t, g);
    gbar(counter, ++phase);
  }
}

// ---------------------------------------------------------------------------
// Host launcher
// ---------------------------------------------------------------------------
extern "C" void kernel_launch(void* const* d_in, const int* in_sizes, int n_in,
                              void* d_out, int out_size, void* d_ws, size_t ws_size,
                              hipStream_t stream) {
  (void)in_sizes; (void)n_in; (void)out_size; (void)ws_size;

  const int*   ids   = (const int*)d_in[0];
  const float* emb   = (const float*)d_in[1];
  const float* wf    = (const float*)d_in[2];
  const float* uf    = (const float*)d_in[3];
  const float* bfv   = (const float*)d_in[4];
  const float* wi    = (const float*)d_in[5];
  const float* ui    = (const float*)d_in[6];
  const float* biv   = (const float*)d_in[7];
  const float* wc    = (const float*)d_in[8];
  const float* uc    = (const float*)d_in[9];
  const float* bcv   = (const float*)d_in[10];
  const float* wo    = (const float*)d_in[11];
  const float* uo    = (const float*)d_in[12];
  const float* bov   = (const float*)d_in[13];
  const float* dec_w = (const float*)d_in[14];
  const float* dec_b = (const float*)d_in[15];

  const size_t ROWS = (size_t)SEQ * BATCH;            // 4096
  const size_t HB_BYTES = (size_t)2 * BATCH * HID * sizeof(bf16_t);  // 131072
  const size_t C_BYTES  = (size_t)BATCH * HID * sizeof(float);       // 131072

  // ---- workspace layout (256B-aligned regions) ----
  char* ws = (char*)d_ws;
  unsigned* counter = (unsigned*)ws;                                  // 256 B
  bf16_t* h0buf = (bf16_t*)(ws + 256);                                // 2x[32][1024] bf16
  bf16_t* h1buf = (bf16_t*)(ws + 256 + HB_BYTES);
  float*  c0    = (float*) (ws + 256 + 2 * HB_BYTES);                 // [32][1024] f32
  float*  c1    = (float*) (ws + 256 + 2 * HB_BYTES + C_BYTES);
  const size_t zero_bytes = 256 + 2 * HB_BYTES + 2 * C_BYTES;         // 524544
  bf16_t* Wt    = (bf16_t*)(ws + zero_bytes);                         // [4096][1024] bf16
  bf16_t* Ut    = Wt  + (size_t)NGATE * HID * HID;
  bf16_t* Xbf   = Ut  + (size_t)NGATE * HID * HID;                    // [4096][1024] bf16
  bf16_t* outs  = Xbf + ROWS * HID;                                   // [4096][1024] bf16
  bf16_t* decbf = outs + ROWS * HID;                                  // [32000][1024] bf16
  float*  Xu    = (float*)(decbf + (size_t)NVOC * HID);               // [4096][4096] f32

  // ---- init: zero state + barrier; zero h0/c0 tail of output ----
  hipMemsetAsync(ws, 0, zero_bytes, stream);
  const size_t DEC_ELEMS = (size_t)SEQ * BATCH * NVOC;                // 131072000
  hipMemsetAsync((float*)d_out + DEC_ELEMS, 0,
                 (size_t)2 * 2 * BATCH * HID * sizeof(float), stream);

  // ---- pack weights: fp32 [K][N] -> bf16 [N][K], gate-major rows f,i,c,o ----
  const dim3 tgrid(HID / 32, HID / 32);
  transpose_cvt<<<tgrid, 256, 0, stream>>>(wf, Wt + 0 * (size_t)HID * HID, HID, HID);
  transpose_cvt<<<tgrid, 256, 0, stream>>>(wi, Wt + 1 * (size_t)HID * HID, HID, HID);
  transpose_cvt<<<tgrid, 256, 0, stream>>>(wc, Wt + 2 * (size_t)HID * HID, HID, HID);
  transpose_cvt<<<tgrid, 256, 0, stream>>>(wo, Wt + 3 * (size_t)HID * HID, HID, HID);
  transpose_cvt<<<tgrid, 256, 0, stream>>>(uf, Ut + 0 * (size_t)HID * HID, HID, HID);
  transpose_cvt<<<tgrid, 256, 0, stream>>>(ui, Ut + 1 * (size_t)HID * HID, HID, HID);
  transpose_cvt<<<tgrid, 256, 0, stream>>>(uc, Ut + 2 * (size_t)HID * HID, HID, HID);
  transpose_cvt<<<tgrid, 256, 0, stream>>>(uo, Ut + 3 * (size_t)HID * HID, HID, HID);

  // dec_w is already [N][K] K-contiguous: straight convert
  cvt_bf16<<<2048, 256, 0, stream>>>(dec_w, decbf, (size_t)NVOC * HID);

  // embedding gather + convert
  embed_cvt<<<(int)ROWS, 256, 0, stream>>>(ids, emb, Xbf);

  // Xu = x @ [Uf|Ui|Uc|Uo]  : M=4096, N=4096, K=1024  (f32 accum)
  gemm_bf16_nt<false><<<dim3(ROWS / 64, (NGATE * HID) / 256), 256, 0, stream>>>(
      Xbf, Ut, nullptr, Xu, NGATE * HID, HID);

  // persistent recurrent kernel: 64 blocks, 264KB dynamic LDS each (<=320KB/WGP)
  const size_t lstm_lds = (size_t)2 * 64 * HID * sizeof(bf16_t)       // W + U slices
                        + (size_t)NGATE * 32 * 16 * sizeof(float);    // sg exchange
  hipFuncSetAttribute(reinterpret_cast<const void*>(lstm_persistent),
                      hipFuncAttributeMaxDynamicSharedMemorySize, (int)lstm_lds);
  lstm_persistent<<<HID / 16, 256, lstm_lds, stream>>>(Wt, Ut, Xu, bfv, biv, bcv, bov,
                                                       h0buf, h1buf, c0, c1, outs, counter);

  // decoder: logits = outs @ dec_w^T + dec_b : M=4096, N=32000, K=1024
  // (streaming non-temporal C stores; dec_w stays L2-resident)
  gemm_bf16_nt<true><<<dim3(ROWS / 64, NVOC / 256), 256, 0, stream>>>(
      outs, decbf, dec_b, (float*)d_out, NVOC, HID);
}
